// GatedAttention_48928267436375
// MI455X (gfx1250) — matrix operand order repaired and checked by reading, not compiled, not run on hardware
//
#include <hip/hip_runtime.h>
#include <math.h>

// ---------------------------------------------------------------------------
// Types / helpers
// ---------------------------------------------------------------------------
typedef unsigned short u16;
typedef unsigned int   u32;

typedef __bf16 bf16;
typedef bf16  v16bf __attribute__((ext_vector_type(16)));
typedef float v8f   __attribute__((ext_vector_type(8)));

union FragBF { v16bf v; u32 u[8]; };

__device__ __forceinline__ float bf2f(u16 h) {
    union { u32 u; float f; } x; x.u = ((u32)h) << 16; return x.f;
}
__device__ __forceinline__ u16 f2bf(float f) {
    union { float f; u32 u; } x; x.f = f;
    u32 r = (x.u >> 16) & 1u;
    x.u += 0x7fffu + r;
    return (u16)(x.u >> 16);
}

// Async global -> LDS copy of 16 bytes (per lane), tracked by ASYNCcnt.
__device__ __forceinline__ void async_copy_b128(const u16* gaddr, void* lds_ptr) {
    u32 ldsoff = (u32)(unsigned long long)lds_ptr;   // addrspace(3) offset
    asm volatile("global_load_async_to_lds_b128 %0, %1, off"
                 :: "v"(ldsoff), "v"(gaddr) : "memory");
}
__device__ __forceinline__ void async_wait0() {
    asm volatile("s_wait_asynccnt 0x0" ::: "memory");
}
__device__ __forceinline__ void async_wait2() {
    asm volatile("s_wait_asynccnt 0x2" ::: "memory");
}

#define Bn   4
#define Sdim 2048
#define Ddim 2048
#define Hn   16
#define Gn   4
#define HDn  128
#define LOG2E 1.4426950408889634f

// ---------------------------------------------------------------------------
// Elementwise f32 -> bf16 cast
// ---------------------------------------------------------------------------
__global__ __launch_bounds__(256) void cast_f32_bf16(
    const float* __restrict__ src, u16* __restrict__ dst, int n4)
{
    int i = blockIdx.x * blockDim.x + threadIdx.x;
    int stride = gridDim.x * blockDim.x;
    for (; i < n4; i += stride) {
        float4 v = ((const float4*)src)[i];
        u16 o0 = f2bf(v.x), o1 = f2bf(v.y), o2 = f2bf(v.z), o3 = f2bf(v.w);
        u32 lo = (u32)o0 | ((u32)o1 << 16);
        u32 hi = (u32)o2 | ((u32)o3 << 16);
        ((uint2*)dst)[i] = make_uint2(lo, hi);
    }
}

// ---------------------------------------------------------------------------
// Generic bf16 WMMA GEMM: C[M,N] = A[M,K] * B[K,N]
//   128x128 block tile, K-step 32, 8 waves each computing a 32x64 sub-tile.
//   Double-buffered software pipeline: A tile via async global->LDS DMA
//   (s_wait_asynccnt-throttled), B tile prefetched into registers, then
//   transposed into LDS.  One barrier per k-step.
//   mode 0: store bf16   mode 1: sigmoid -> bf16   mode 2: store f32
// ---------------------------------------------------------------------------
__global__ __launch_bounds__(256) void gemm_bf16_wmma(
    const u16* __restrict__ A, const u16* __restrict__ Bw,
    u16* __restrict__ Cb, float* __restrict__ Cf,
    int M, int N, int K, int mode)
{
    __shared__ u16 As[2][128][32];   // [m][k]
    __shared__ u16 Bs[2][128][32];   // [n][k]  (transposed on commit)

    const int tid  = threadIdx.x;
    const int lane = tid & 31, wave = tid >> 5;
    const int half = lane >> 4, l = lane & 15;
    const int bm = blockIdx.y * 128, bn = blockIdx.x * 128;
    const int wm = (wave >> 1) * 32, wn = (wave & 1) * 64;

    // loop-invariant load coordinates
    const int a_row0 = (tid * 2) >> 2,        a_q0 = ((tid * 2) & 3) * 8;
    const int a_row1 = (tid * 2 + 1) >> 2,    a_q1 = ((tid * 2 + 1) & 3) * 8;
    const int b_kr0  = (tid * 2) >> 4,        b_n0 = ((tid * 2) & 15) * 8;
    const int b_kr1  = (tid * 2 + 1) >> 4,    b_n1 = ((tid * 2 + 1) & 15) * 8;

    v8f acc[2][4];
    const v8f vzero = {0.f,0.f,0.f,0.f,0.f,0.f,0.f,0.f};
#pragma unroll
    for (int i = 0; i < 2; i++)
#pragma unroll
        for (int j = 0; j < 4; j++) acc[i][j] = vzero;

    const int niter = K >> 5;
    uint4 breg0, breg1;

    // ---- prologue: stage tile 0 ----
    async_copy_b128(A + (size_t)(bm + a_row0) * K + a_q0, &As[0][a_row0][a_q0]);
    async_copy_b128(A + (size_t)(bm + a_row1) * K + a_q1, &As[0][a_row1][a_q1]);
    breg0 = *(const uint4*)(Bw + (size_t)b_kr0 * N + bn + b_n0);
    breg1 = *(const uint4*)(Bw + (size_t)b_kr1 * N + bn + b_n1);

    for (int it = 0; it < niter; ++it) {
        const int buf = it & 1;
        // ---- commit B registers to LDS (transposed) ----
        {
            const u16* e0 = (const u16*)&breg0;
            const u16* e1 = (const u16*)&breg1;
#pragma unroll
            for (int j = 0; j < 8; j++) Bs[buf][b_n0 + j][b_kr0] = e0[j];
#pragma unroll
            for (int j = 0; j < 8; j++) Bs[buf][b_n1 + j][b_kr1] = e1[j];
        }
        // ---- prefetch tile it+1 ----
        if (it + 1 < niter) {
            const int kt = (it + 1) << 5;
            async_copy_b128(A + (size_t)(bm + a_row0) * K + kt + a_q0,
                            &As[buf ^ 1][a_row0][a_q0]);
            async_copy_b128(A + (size_t)(bm + a_row1) * K + kt + a_q1,
                            &As[buf ^ 1][a_row1][a_q1]);
            breg0 = *(const uint4*)(Bw + (size_t)(kt + b_kr0) * N + bn + b_n0);
            breg1 = *(const uint4*)(Bw + (size_t)(kt + b_kr1) * N + bn + b_n1);
            async_wait2();    // in-order: current tile's 2 copies have landed
        } else {
            async_wait0();
        }
        __syncthreads();

        // ---- fragments + WMMA ----
        FragBF a[2], bfr[4];
#pragma unroll
        for (int mf = 0; mf < 2; mf++) {
            const u32* p = (const u32*)&As[buf][wm + mf * 16 + l][0];
#pragma unroll
            for (int j = 0; j < 4; j++) {
                a[mf].u[j]     = p[half * 4 + j];       // K = half*8 + 2j
                a[mf].u[j + 4] = p[8 + half * 4 + j];   // K = 16 + half*8 + 2j
            }
        }
#pragma unroll
        for (int nf = 0; nf < 4; nf++) {
            const u32* p = (const u32*)&Bs[buf][wn + nf * 16 + l][0];
#pragma unroll
            for (int j = 0; j < 4; j++) {
                bfr[nf].u[j]     = p[half * 4 + j];       // K = half*8 + 2j
                bfr[nf].u[j + 4] = p[8 + half * 4 + j];   // K = 16 + half*8 + 2j
            }
        }
#pragma unroll
        for (int mf = 0; mf < 2; mf++)
#pragma unroll
            for (int nf = 0; nf < 4; nf++)
                acc[mf][nf] = __builtin_amdgcn_wmma_f32_16x16x32_bf16(
                    false, a[mf].v, false, bfr[nf].v, (short)0, acc[mf][nf],
                    false, false);
    }

    // ---- epilogue (mode branch hoisted out of the store loops) ----
    const int rbase = bm + wm + half * 8;
    const int cbase = bn + wn + l;
    if (mode == 2) {
#pragma unroll
        for (int mf = 0; mf < 2; mf++)
#pragma unroll
            for (int nf = 0; nf < 4; nf++)
#pragma unroll
                for (int e = 0; e < 8; e++)
                    Cf[(size_t)(rbase + mf * 16 + e) * N + cbase + nf * 16] =
                        acc[mf][nf][e];
    } else if (mode == 1) {
#pragma unroll
        for (int mf = 0; mf < 2; mf++)
#pragma unroll
            for (int nf = 0; nf < 4; nf++)
#pragma unroll
                for (int e = 0; e < 8; e++) {
                    float v = acc[mf][nf][e];
                    float s = __builtin_amdgcn_rcpf(1.0f + exp2f(-v * LOG2E));
                    Cb[(size_t)(rbase + mf * 16 + e) * N + cbase + nf * 16] = f2bf(s);
                }
    } else {
#pragma unroll
        for (int mf = 0; mf < 2; mf++)
#pragma unroll
            for (int nf = 0; nf < 4; nf++)
#pragma unroll
                for (int e = 0; e < 8; e++)
                    Cb[(size_t)(rbase + mf * 16 + e) * N + cbase + nf * 16] =
                        f2bf(acc[mf][nf][e]);
    }
}

// ---------------------------------------------------------------------------
// Zero-centered RMSNorm + RoPE over bf16 head rows, in place.
// One wave per (b,s,head) row of 128; lane owns d, d+32, d+64, d+96.
// ---------------------------------------------------------------------------
__global__ __launch_bounds__(32) void rmsnorm_rope(
    u16* __restrict__ buf, const float* __restrict__ scale,
    const float* __restrict__ cosb, const float* __restrict__ sinb, int heads)
{
    int idx = blockIdx.x;
    int s   = (idx / heads) % Sdim;
    size_t base = (size_t)idx * HDn;
    int l = threadIdx.x;

    float x0 = bf2f(buf[base + l]);
    float x1 = bf2f(buf[base + l + 32]);
    float x2 = bf2f(buf[base + l + 64]);
    float x3 = bf2f(buf[base + l + 96]);

    float ss = x0 * x0 + x1 * x1 + x2 * x2 + x3 * x3;
#pragma unroll
    for (int off = 1; off < 32; off <<= 1) ss += __shfl_xor(ss, off);
    float rms = rsqrtf(ss * (1.0f / 128.0f) + 1e-6f);

    float y0 = x0 * rms * (1.0f + scale[l]);
    float y1 = x1 * rms * (1.0f + scale[l + 32]);
    float y2 = x2 * rms * (1.0f + scale[l + 64]);
    float y3 = x3 * rms * (1.0f + scale[l + 96]);

    const float* cs = cosb + (size_t)s * HDn;
    const float* sn = sinb + (size_t)s * HDn;
    // rotate-half: out[d<64] = y*cos - y[d+64]*sin ; out[d>=64] = y*cos + y[d-64]*sin
    float o0 = y0 * cs[l]      - y2 * sn[l];
    float o1 = y1 * cs[l + 32] - y3 * sn[l + 32];
    float o2 = y2 * cs[l + 64] + y0 * sn[l + 64];
    float o3 = y3 * cs[l + 96] + y1 * sn[l + 96];

    buf[base + l]      = f2bf(o0);
    buf[base + l + 32] = f2bf(o1);
    buf[base + l + 64] = f2bf(o2);
    buf[base + l + 96] = f2bf(o3);
}

// ---------------------------------------------------------------------------
// Causal GQA flash attention, bf16 WMMA, online softmax (log2 domain),
// fused sigmoid-gate multiply.  Block = (qblock 128, head, batch); 256
// threads = 8 waves, each wave owns 16 query rows.  Q/K tiles via async
// global->LDS DMA; V tile prefetched to registers during QK^T + softmax.
// ---------------------------------------------------------------------------
__global__ __launch_bounds__(256) void flash_attn_wmma(
    const u16* __restrict__ Q, const u16* __restrict__ Kb,
    const u16* __restrict__ Vb, const u16* __restrict__ Gt,
    u16* __restrict__ Out)
{
    __shared__ u16 Qs[128][128];        // [qrow][hd]      32 KB
    __shared__ u16 KVs[128][128];       // K:[key][hd] then V^T:[hd][key] 32 KB
    __shared__ u16 Ps[8][16][128];      // per-wave P staging              32 KB

    const int tid  = threadIdx.x;
    const int lane = tid & 31, wave = tid >> 5;
    const int half = lane >> 4, l = lane & 15;
    const int qb = blockIdx.x, h = blockIdx.y, b = blockIdx.z;
    const int g  = h / (Hn / Gn);
    const float sc2 = 0.08838834764831845f * LOG2E;  // 1/sqrt(128) * log2(e)
    const v8f vzero = {0.f,0.f,0.f,0.f,0.f,0.f,0.f,0.f};

    // ---- load Q tile (async DMA) ----
#pragma unroll
    for (int i = 0; i < 8; i++) {
        int id = tid * 8 + i;                 // 0..2047
        int r = id >> 4, c = (id & 15) * 8;
        int s = qb * 128 + r;
        async_copy_b128(Q + (((size_t)b * Sdim + s) * Hn + h) * HDn + c, &Qs[r][c]);
    }

    float mrow[8], lrow[8];
    v8f oacc[8];
#pragma unroll
    for (int e = 0; e < 8; e++) { mrow[e] = -3.0e38f; lrow[e] = 0.0f; }
#pragma unroll
    for (int nt = 0; nt < 8; nt++) oacc[nt] = vzero;

    for (int kb = 0; kb <= qb; kb++) {
        __syncthreads();                       // prev PV reads done
        // ---- K tile: KVs[key][hd] (async DMA); V tile -> registers ----
        uint4 vreg[8];
#pragma unroll
        for (int i = 0; i < 8; i++) {
            int id = tid * 8 + i;
            int r = id >> 4, c = (id & 15) * 8;
            int key = kb * 128 + r;
            async_copy_b128(Kb + (((size_t)b * Sdim + key) * Gn + g) * HDn + c,
                            &KVs[r][c]);
            vreg[i] = *(const uint4*)(Vb + (((size_t)b * Sdim + key) * Gn + g) * HDn + c);
        }
        async_wait0();                          // Q (first iter) + K landed
        __syncthreads();

        // ---- scores: S = Q * K^T  (16 x 128 per wave) ----
        v8f sacc[8];
#pragma unroll
        for (int nt = 0; nt < 8; nt++) sacc[nt] = vzero;
#pragma unroll
        for (int ch = 0; ch < 4; ch++) {
            FragBF a;
            const u32* pa = (const u32*)&Qs[wave * 16 + l][ch * 32];
#pragma unroll
            for (int j = 0; j < 4; j++) {
                a.u[j]     = pa[half * 4 + j];
                a.u[j + 4] = pa[8 + half * 4 + j];
            }
#pragma unroll
            for (int nt = 0; nt < 8; nt++) {
                FragBF bf;
                const u32* pb = (const u32*)&KVs[nt * 16 + l][ch * 32];
#pragma unroll
                for (int j = 0; j < 4; j++) {
                    bf.u[j]     = pb[half * 4 + j];
                    bf.u[j + 4] = pb[8 + half * 4 + j];
                }
                sacc[nt] = __builtin_amdgcn_wmma_f32_16x16x32_bf16(
                    false, a.v, false, bf.v, (short)0, sacc[nt], false, false);
            }
        }

        // ---- scale + causal mask + online softmax (log2 domain) ----
#pragma unroll
        for (int e = 0; e < 8; e++) {
            int q_g = qb * 128 + wave * 16 + e + half * 8;
            float tm = -3.0e38f;
#pragma unroll
            for (int nt = 0; nt < 8; nt++) {
                float v = sacc[nt][e] * sc2;
                int k_g = kb * 128 + nt * 16 + l;
                if (k_g > q_g) v = -1.0e30f;
                sacc[nt][e] = v;
                tm = fmaxf(tm, v);
            }
#pragma unroll
            for (int off = 1; off < 16; off <<= 1)
                tm = fmaxf(tm, __shfl_xor(tm, off));
            float mnew  = fmaxf(mrow[e], tm);
            float alpha = exp2f(mrow[e] - mnew);
            float rs = 0.0f;
            int prow = e + half * 8;
#pragma unroll
            for (int nt = 0; nt < 8; nt++) {
                float pv = exp2f(sacc[nt][e] - mnew);
                rs += pv;
                Ps[wave][prow][nt * 16 + l] = f2bf(pv);
            }
#pragma unroll
            for (int off = 1; off < 16; off <<= 1) rs += __shfl_xor(rs, off);
            lrow[e] = lrow[e] * alpha + rs;
            mrow[e] = mnew;
#pragma unroll
            for (int nt = 0; nt < 8; nt++) oacc[nt][e] *= alpha;
        }
        __syncthreads();                       // Ks consumed, Ps published

        // ---- commit V tile transposed: KVs[hd][key] ----
#pragma unroll
        for (int i = 0; i < 8; i++) {
            int id = tid * 8 + i;
            int r = id >> 4, c = (id & 15) * 8;
            const u16* ep = (const u16*)&vreg[i];
#pragma unroll
            for (int j = 0; j < 8; j++) KVs[c + j][r] = ep[j];
        }
        __syncthreads();

        // ---- O += P * V ----
#pragma unroll
        for (int ch = 0; ch < 4; ch++) {
            FragBF a;
            const u32* pa = (const u32*)&Ps[wave][l][ch * 32];
#pragma unroll
            for (int j = 0; j < 4; j++) {
                a.u[j]     = pa[half * 4 + j];
                a.u[j + 4] = pa[8 + half * 4 + j];
            }
#pragma unroll
            for (int nt = 0; nt < 8; nt++) {
                FragBF bf;
                const u32* pb = (const u32*)&KVs[nt * 16 + l][ch * 32];
#pragma unroll
                for (int j = 0; j < 4; j++) {
                    bf.u[j]     = pb[half * 4 + j];
                    bf.u[j + 4] = pb[8 + half * 4 + j];
                }
                oacc[nt] = __builtin_amdgcn_wmma_f32_16x16x32_bf16(
                    false, a.v, false, bf.v, (short)0, oacc[nt], false, false);
            }
        }
    }

    // ---- epilogue: normalize (rcp), multiply sigmoid gate, store bf16 ----
    float rl[8];
#pragma unroll
    for (int e = 0; e < 8; e++) rl[e] = __builtin_amdgcn_rcpf(lrow[e]);
#pragma unroll
    for (int nt = 0; nt < 8; nt++)
#pragma unroll
        for (int e = 0; e < 8; e++) {
            int srow = qb * 128 + wave * 16 + e + half * 8;
            int hd   = nt * 16 + l;
            size_t idx = ((size_t)b * Sdim + srow) * (Hn * HDn) + h * HDn + hd;
            float o = oacc[nt][e] * rl[e];
            Out[idx] = f2bf(o * bf2f(Gt[idx]));
        }
}

// ---------------------------------------------------------------------------
// Host-side launch
// ---------------------------------------------------------------------------
extern "C" void kernel_launch(void* const* d_in, const int* in_sizes, int n_in,
                              void* d_out, int out_size, void* d_ws, size_t ws_size,
                              hipStream_t stream)
{
    (void)in_sizes; (void)n_in; (void)out_size; (void)ws_size;
    const float* x       = (const float*)d_in[0];
    // d_in[1] = mask (causal, recomputed analytically)
    const float* cosb    = (const float*)d_in[2];
    const float* sinb    = (const float*)d_in[3];
    const float* wq      = (const float*)d_in[4];
    const float* wk      = (const float*)d_in[5];
    const float* wv      = (const float*)d_in[6];
    const float* wg      = (const float*)d_in[7];
    const float* wo      = (const float*)d_in[8];
    const float* q_scale = (const float*)d_in[9];
    const float* k_scale = (const float*)d_in[10];
    float* out = (float*)d_out;

    u16* ws = (u16*)d_ws;
    const size_t NX = (size_t)Bn * Sdim * Ddim;      // 16,777,216
    u16* xb  = ws;
    u16* wqb = xb  + NX;
    u16* wkb = wqb + (size_t)Ddim * Hn * HDn;        // 4,194,304
    u16* wvb = wkb + (size_t)Ddim * Gn * HDn;        // 1,048,576
    u16* wgb = wvb + (size_t)Ddim * Gn * HDn;
    u16* wob = wgb + (size_t)Ddim * Hn * HDn;
    u16* qbf = wob + (size_t)Hn * HDn * Ddim;
    u16* kbf = qbf + NX;
    u16* vbf = kbf + (size_t)Bn * Sdim * Gn * HDn;
    u16* gtb = vbf + (size_t)Bn * Sdim * Gn * HDn;
    u16* cgb = gtb + NX;

    // 1) casts to bf16
    cast_f32_bf16<<<4096, 256, 0, stream>>>(x,  xb,  (int)(NX / 4));
    cast_f32_bf16<<<2048, 256, 0, stream>>>(wq, wqb, 4194304 / 4);
    cast_f32_bf16<<<1024, 256, 0, stream>>>(wk, wkb, 1048576 / 4);
    cast_f32_bf16<<<1024, 256, 0, stream>>>(wv, wvb, 1048576 / 4);
    cast_f32_bf16<<<2048, 256, 0, stream>>>(wg, wgb, 4194304 / 4);
    cast_f32_bf16<<<2048, 256, 0, stream>>>(wo, wob, 4194304 / 4);

    // 2) projections (M = B*S = 8192, K = 2048)
    dim3 blk(256);
    gemm_bf16_wmma<<<dim3(16, 64), blk, 0, stream>>>(xb, wqb, qbf, nullptr, 8192, 2048, 2048, 0);
    gemm_bf16_wmma<<<dim3( 4, 64), blk, 0, stream>>>(xb, wkb, kbf, nullptr, 8192,  512, 2048, 0);
    gemm_bf16_wmma<<<dim3( 4, 64), blk, 0, stream>>>(xb, wvb, vbf, nullptr, 8192,  512, 2048, 0);
    gemm_bf16_wmma<<<dim3(16, 64), blk, 0, stream>>>(xb, wgb, gtb, nullptr, 8192, 2048, 2048, 1);

    // 3) zero-centered RMSNorm + RoPE on q and k
    rmsnorm_rope<<<Bn * Sdim * Hn, 32, 0, stream>>>(qbf, q_scale, cosb, sinb, Hn);
    rmsnorm_rope<<<Bn * Sdim * Gn, 32, 0, stream>>>(kbf, k_scale, cosb, sinb, Gn);

    // 4) causal GQA flash attention + gate
    flash_attn_wmma<<<dim3(Sdim / 128, Hn, Bn), blk, 0, stream>>>(qbf, kbf, vbf, gtb, cgb);

    // 5) output projection -> f32
    gemm_bf16_wmma<<<dim3(16, 64), blk, 0, stream>>>(cgb, wob, nullptr, out, 8192, 2048, 2048, 2);
}
